// DLRM_Net_21045339750931
// MI455X (gfx1250) — matrix operand order repaired and checked by reading, not compiled, not run on hardware
//
#include <hip/hip_runtime.h>
#include <hip/hip_fp16.h>
#include <cstdint>

typedef _Float16 half_t;
typedef __attribute__((ext_vector_type(8)))  _Float16 v8h;
typedef __attribute__((ext_vector_type(16))) _Float16 v16h;
typedef __attribute__((ext_vector_type(8)))  float    v8f;
typedef __attribute__((ext_vector_type(2)))  _Float16 h2;
typedef __attribute__((ext_vector_type(4)))  int      v4i;

// exact parameter types for __builtin_amdgcn_global_load_async_to_lds_b128:
//   arg0: int4 addrspace(1)* (global, printed as "int4 __device__*")
//   arg1: int4 addrspace(3)* (LDS)
#define GLB_V4(p) ((__attribute__((address_space(1))) v4i*)(p))
#define LDS_V4(p) ((__attribute__((address_space(3))) v4i*)(p))

#define BATCH   8192
#define NTAB    26
#define POOL    10
#define DEMB    64
#define EROWS   100000
#define ZDIM    1728   // 64 + 26*64

// Async global->LDS path (CDNA5): probe the builtin on the device pass only.
#if defined(__HIP_DEVICE_COMPILE__) && __has_builtin(__builtin_amdgcn_global_load_async_to_lds_b128)
#define HAVE_ASYNC_LDS 1
#else
#define HAVE_ASYNC_LDS 0
#endif

#if defined(__HIP_DEVICE_COMPILE__) && __has_builtin(__builtin_amdgcn_s_wait_asynccnt)
#define WAIT_ASYNC() __builtin_amdgcn_s_wait_asynccnt(0)
#else
#define WAIT_ASYNC() asm volatile("s_wait_asynccnt 0x0" ::: "memory")
#endif

// ---------------------------------------------------------------------------
// f32 -> f16 convert with row/col zero padding (weights + dense input)
// ---------------------------------------------------------------------------
__global__ void cvt_pad(const float* __restrict__ src, half_t* __restrict__ dst,
                        int dstRows, int srcRows, int K, int Kp) {
    int idx = blockIdx.x * blockDim.x + threadIdx.x;
    int total = dstRows * Kp;
    if (idx >= total) return;
    int r = idx / Kp;
    int k = idx - r * Kp;
    float v = (r < srcRows && k < K) ? src[(size_t)r * K + k] : 0.0f;
    dst[idx] = (half_t)v;
}

// ---------------------------------------------------------------------------
// EmbeddingBag(sum): one wave32 per bag, float2 per lane covers the 256B row.
// 545MB of random HBM reads -> pure bandwidth kernel.  Result written as f16
// straight into the concat buffer Z at col 64 + t*64.
// ---------------------------------------------------------------------------
__global__ __launch_bounds__(256) void embed_pool(
        const int*   __restrict__ indices,   // [NTAB, BATCH*POOL]
        const float* __restrict__ tables,    // [NTAB, EROWS, DEMB]
        half_t*      __restrict__ Z) {       // [BATCH, ZDIM] f16
    int tid  = threadIdx.x;
    int lane = tid & 31;
    int wave = tid >> 5;
    int bag  = blockIdx.x * 8 + wave;        // 8 waves per block
    if (bag >= NTAB * BATCH) return;
    int t = bag / BATCH;
    int b = bag - t * BATCH;

    const float* tab  = tables + (size_t)t * EROWS * DEMB;
    const int*   idxp = indices + (size_t)t * (BATCH * POOL) + (size_t)b * POOL;

    float ax = 0.0f, ay = 0.0f;
#pragma unroll
    for (int l = 0; l < POOL; ++l) {
        int row = idxp[l];
        const float2 v = *(const float2*)(tab + (size_t)row * DEMB + lane * 2);
        ax += v.x;
        ay += v.y;
    }
    h2 o;
    o.x = (half_t)ax;
    o.y = (half_t)ay;
    *(h2*)(Z + (size_t)b * ZDIM + 64 + t * DEMB + lane * 2) = o;
}

// ---------------------------------------------------------------------------
// WMMA GEMM:  C[M,N] = act(X[M,K] @ W[N,K]^T + bias)
//   f16 inputs, f32 accum via v_wmma_f32_16x16x32_f16.
//   128 threads = 4 waves; tile 128(M) x 64(N); K-step 32.
//   Double-buffered LDS; next tile staged with async global->LDS copies
//   (ASYNCcnt) while the current tile's 32 WMMAs execute; one barrier/step.
//   MODE 0: ReLU -> f16 C16;  MODE 1: sigmoid -> f32 Cf (cols < n_valid)
// ---------------------------------------------------------------------------
#define TILE_M 128
#define TILE_N 64
#define TILE_K 32
#define LSTR   40   // LDS row stride in halves (80B): conflict-free b128 reads

template<int MODE>
__global__ __launch_bounds__(128) void gemm_wmma(
        const half_t* __restrict__ X, int ldx,
        const half_t* __restrict__ W, int ldw,
        const float*  __restrict__ bias,
        half_t* __restrict__ C16, float* __restrict__ Cf,
        int N, int K, int ldc, int n_valid) {
    __shared__ alignas(16) half_t Xs[2][TILE_M * LSTR];
    __shared__ alignas(16) half_t Ws[2][TILE_N * LSTR];

    const int tid   = threadIdx.x;
    const int lane  = tid & 31;
    const int wave  = tid >> 5;
    const int l16   = lane & 15;
    const int hi    = lane >> 4;
    const int mBase = blockIdx.x * TILE_M;
    const int nBase = blockIdx.y * TILE_N;

    // staging map: 4 lanes cover one contiguous 64B row segment
    const int xrow   = tid >> 2;          // 0..31 (+ q*32)
    const int xcol   = (tid & 3) * 8;     // half offset 0/8/16/24
    const int wrow   = tid >> 1;          // 0..63
    const int wcol   = (tid & 1) * 16;    // half offset 0/16
    const int gn     = nBase + wrow;

    // pre-zero padded W rows once; staging skips them so zeros persist
    if (gn >= N) {
        v8h z = {};
#pragma unroll
        for (int b = 0; b < 2; ++b) {
            *(v8h*)(&Ws[b][wrow * LSTR + wcol])     = z;
            *(v8h*)(&Ws[b][wrow * LSTR + wcol + 8]) = z;
        }
    }

    v8f acc[2][4];
#pragma unroll
    for (int mi = 0; mi < 2; ++mi)
#pragma unroll
        for (int ni = 0; ni < 4; ++ni) {
            v8f z = {0.f, 0.f, 0.f, 0.f, 0.f, 0.f, 0.f, 0.f};
            acc[mi][ni] = z;
        }

    auto compute = [&](int buf) {
        v16h afrag[2], bfrag[4];
#pragma unroll
        for (int mi = 0; mi < 2; ++mi) {
            const half_t* p = &Xs[buf][(wave * 32 + mi * 16 + l16) * LSTR + hi * 8];
            v8h lo = *(const v8h*)(p);
            v8h hh = *(const v8h*)(p + 16);
            afrag[mi] = __builtin_shufflevector(lo, hh,
                0,1,2,3,4,5,6,7,8,9,10,11,12,13,14,15);
        }
#pragma unroll
        for (int ni = 0; ni < 4; ++ni) {
            const half_t* p = &Ws[buf][(ni * 16 + l16) * LSTR + hi * 8];
            v8h lo = *(const v8h*)(p);
            v8h hh = *(const v8h*)(p + 16);
            bfrag[ni] = __builtin_shufflevector(lo, hh,
                0,1,2,3,4,5,6,7,8,9,10,11,12,13,14,15);
        }
#pragma unroll
        for (int mi = 0; mi < 2; ++mi)
#pragma unroll
            for (int ni = 0; ni < 4; ++ni)
                acc[mi][ni] = __builtin_amdgcn_wmma_f32_16x16x32_f16(
                    false, afrag[mi], false, bfrag[ni],
                    (short)0, acc[mi][ni], false, false);
    };

    const int ntiles = K / TILE_K;

#if HAVE_ASYNC_LDS
    auto stage = [&](int buf, int kk) {
#pragma unroll
        for (int q = 0; q < 4; ++q) {
            const half_t* gs = X + (size_t)(mBase + xrow + q * 32) * ldx + kk + xcol;
            half_t*       ld = &Xs[buf][(xrow + q * 32) * LSTR + xcol];
            __builtin_amdgcn_global_load_async_to_lds_b128(GLB_V4(gs), LDS_V4(ld), 0, 0);
        }
        if (gn < N) {
            const half_t* gs = W + (size_t)gn * ldw + kk + wcol;
            half_t*       ld = &Ws[buf][wrow * LSTR + wcol];
            __builtin_amdgcn_global_load_async_to_lds_b128(GLB_V4(gs),     LDS_V4(ld),     0, 0);
            __builtin_amdgcn_global_load_async_to_lds_b128(GLB_V4(gs + 8), LDS_V4(ld + 8), 0, 0);
        }
    };

    stage(0, 0);
    WAIT_ASYNC();
    __syncthreads();
    for (int i = 0; i < ntiles; ++i) {
        const int cur = i & 1;
        if (i + 1 < ntiles) stage(cur ^ 1, (i + 1) * TILE_K);
        compute(cur);
        if (i + 1 < ntiles) {
            WAIT_ASYNC();
            __syncthreads();
        }
    }
#else
    // fallback: register-staged software pipeline (double-buffered LDS)
    v8h xr[4], wr[2];
    auto loadRegs = [&](int kk) {
#pragma unroll
        for (int q = 0; q < 4; ++q)
            xr[q] = *(const v8h*)(X + (size_t)(mBase + xrow + q * 32) * ldx + kk + xcol);
        if (gn < N) {
            const half_t* wsp = W + (size_t)gn * ldw + kk + wcol;
            wr[0] = *(const v8h*)(wsp);
            wr[1] = *(const v8h*)(wsp + 8);
        }
    };
    auto storeRegs = [&](int buf) {
#pragma unroll
        for (int q = 0; q < 4; ++q)
            *(v8h*)(&Xs[buf][(xrow + q * 32) * LSTR + xcol]) = xr[q];
        if (gn < N) {
            half_t* wd = &Ws[buf][wrow * LSTR + wcol];
            *(v8h*)(wd)     = wr[0];
            *(v8h*)(wd + 8) = wr[1];
        }
    };

    loadRegs(0);
    storeRegs(0);
    __syncthreads();
    for (int i = 0; i < ntiles; ++i) {
        const int cur = i & 1;
        if (i + 1 < ntiles) loadRegs((i + 1) * TILE_K);   // overlap with WMMAs
        compute(cur);
        if (i + 1 < ntiles) {
            storeRegs(cur ^ 1);
            __syncthreads();
        }
    }
#endif

    // --- epilogue: bias + activation (D layout: m = 8*hi + r)
#pragma unroll
    for (int ni = 0; ni < 4; ++ni) {
        int n = nBase + ni * 16 + l16;
        float bv = (n < n_valid) ? bias[n] : 0.0f;
#pragma unroll
        for (int mi = 0; mi < 2; ++mi) {
            int mrow = mBase + wave * 32 + mi * 16 + hi * 8;
#pragma unroll
            for (int r = 0; r < 8; ++r) {
                float v = acc[mi][ni][r] + bv;
                if (MODE == 0) {
                    v = v > 0.0f ? v : 0.0f;
                    if (n < n_valid)
                        C16[(size_t)(mrow + r) * ldc + n] = (half_t)v;
                } else {
                    v = 1.0f / (1.0f + __expf(-v));
                    if (n < n_valid)
                        Cf[(size_t)(mrow + r) * ldc + n] = v;
                }
            }
        }
    }
}

// ---------------------------------------------------------------------------
// Launcher: prep (f16 conversion) -> bottom MLP -> embeddings -> top MLP
// ---------------------------------------------------------------------------
extern "C" void kernel_launch(void* const* d_in, const int* in_sizes, int n_in,
                              void* d_out, int out_size, void* d_ws, size_t ws_size,
                              hipStream_t stream) {
    (void)in_sizes; (void)n_in; (void)out_size; (void)ws_size;

    const float* dense   = (const float*)d_in[0];
    const int*   indices = (const int*)d_in[1];
    /* d_in[2] = offsets (uniform stride, unused) */
    const float* tables  = (const float*)d_in[3];
    const float* bw0 = (const float*)d_in[4];  const float* bb0 = (const float*)d_in[5];
    const float* bw1 = (const float*)d_in[6];  const float* bb1 = (const float*)d_in[7];
    const float* bw2 = (const float*)d_in[8];  const float* bb2 = (const float*)d_in[9];
    const float* tw0 = (const float*)d_in[10]; const float* tb0 = (const float*)d_in[11];
    const float* tw1 = (const float*)d_in[12]; const float* tb1 = (const float*)d_in[13];
    const float* tw2 = (const float*)d_in[14]; const float* tb2 = (const float*)d_in[15];

    // workspace carve-out (f16 buffers, 256B aligned)
    char* ws = (char*)d_ws;
    size_t off = 0;
    auto alloc = [&](size_t halves) -> half_t* {
        half_t* p = (half_t*)(ws + off);
        off += ((halves * sizeof(half_t)) + 255) & ~(size_t)255;
        return p;
    };
    half_t* X0  = alloc((size_t)BATCH * 32);    // dense padded to K=32
    half_t* X1  = alloc((size_t)BATCH * 512);
    half_t* X2  = alloc((size_t)BATCH * 256);
    half_t* Z   = alloc((size_t)BATCH * ZDIM);  // concat buffer
    half_t* Y1  = alloc((size_t)BATCH * 512);
    half_t* Y2  = alloc((size_t)BATCH * 256);
    half_t* w0h = alloc((size_t)512 * 32);      // bot_w0 [512,13]->[512,32]
    half_t* w1h = alloc((size_t)256 * 512);
    half_t* w2h = alloc((size_t)64  * 256);
    half_t* u0h = alloc((size_t)512 * ZDIM);
    half_t* u1h = alloc((size_t)256 * 512);
    half_t* u2h = alloc((size_t)16  * 256);     // top_w2 [1,256]->[16,256]

    auto cvt = [&](const float* s, half_t* d, int dr, int sr, int k, int kp) {
        int tot = dr * kp;
        cvt_pad<<<dim3((tot + 255) / 256), dim3(256), 0, stream>>>(s, d, dr, sr, k, kp);
    };

    // --- prep: f16 conversions (weights tiny; L2-resident afterwards)
    cvt(dense, X0, BATCH, BATCH, 13, 32);
    cvt(bw0, w0h, 512, 512, 13, 32);
    cvt(bw1, w1h, 256, 256, 512, 512);
    cvt(bw2, w2h, 64, 64, 256, 256);
    cvt(tw0, u0h, 512, 512, ZDIM, ZDIM);
    cvt(tw1, u1h, 256, 256, 512, 512);
    cvt(tw2, u2h, 16, 1, 256, 256);

    const dim3 blk(128);
    const int gM = BATCH / TILE_M;   // 64

    // --- bottom MLP
    gemm_wmma<0><<<dim3(gM, 512 / TILE_N), blk, 0, stream>>>(
        X0, 32, w0h, 32, bb0, X1, nullptr, 512, 32, 512, 512);
    gemm_wmma<0><<<dim3(gM, 256 / TILE_N), blk, 0, stream>>>(
        X1, 512, w1h, 512, bb1, X2, nullptr, 256, 512, 256, 256);
    // layer 2 writes straight into Z cols [0,64)
    gemm_wmma<0><<<dim3(gM, 1), blk, 0, stream>>>(
        X2, 256, w2h, 256, bb2, Z, nullptr, 64, 256, ZDIM, 64);

    // --- embedding bags -> Z cols [64, 1728)
    embed_pool<<<dim3((NTAB * BATCH) / 8), dim3(256), 0, stream>>>(indices, tables, Z);

    // --- top MLP
    gemm_wmma<0><<<dim3(gM, 512 / TILE_N), blk, 0, stream>>>(
        Z, ZDIM, u0h, ZDIM, tb0, Y1, nullptr, 512, ZDIM, 512, 512);
    gemm_wmma<0><<<dim3(gM, 256 / TILE_N), blk, 0, stream>>>(
        Y1, 512, u1h, 512, tb1, Y2, nullptr, 256, 512, 256, 256);
    gemm_wmma<1><<<dim3(gM, 1), blk, 0, stream>>>(
        Y2, 256, u2h, 256, tb2, nullptr, (float*)d_out, 16, 256, 1, 1);
}
